// MoEFuser_80522046865980
// MI455X (gfx1250) — matrix operand order repaired and checked by reading, not compiled, not run on hardware
//
#include <hip/hip_runtime.h>
#include <hip/hip_bf16.h>

// ---------------------------------------------------------------------------
// CDNA5 (gfx1250) fused MoE voxel-fusion pipeline, bf16 WMMA implicit GEMM.
// Padded operands -> branch-free inner loops; 2x cout x 4x voxel blocking.
// ---------------------------------------------------------------------------

typedef __attribute__((ext_vector_type(16))) __bf16 v16bf;
typedef __attribute__((ext_vector_type(8)))  float  v8f;

#define HD 64
#define WD 64
#define LD 8
#define NVOX 32768          // 64*64*8
#define CIN 64
#define COUT 128
#define HID 128
#define NE 3

// padded input (pad 3): 70 x 70 x 14
#define P1H 70
#define P1W 70
#define P1L 14
#define P1VOX (P1H * P1W * P1L)          // 68600
// padded feats (pad 1): 66 x 66 x 10
#define P2H 66
#define P2W 66
#define P2L 10
#define P2VOX (P2H * P2W * P2L)          // 43560
#define P2CENTER ((1 * P2W + 1) * P2L + 1)   // 671

// ushort offset between the two cout-tile A blocks (343 taps * 1024)
#define MT1OFF (343 * 1024)

__device__ __forceinline__ unsigned short f2bf(float f) {
  unsigned int u = __builtin_bit_cast(unsigned int, f);
  u += 0x7FFFu + ((u >> 16) & 1u);           // round-to-nearest-even
  return (unsigned short)(u >> 16);
}
__device__ __forceinline__ float bf2f(unsigned short h) {
  unsigned int u = ((unsigned int)h) << 16;
  return __builtin_bit_cast(float, u);
}
__device__ __forceinline__ v8f wmma_bf16(v16bf a, v16bf b, v8f c) {
  return __builtin_amdgcn_wmma_f32_16x16x32_bf16(false, a, false, b,
                                                 (short)0, c, false, false);
}

// ---------------------------------------------------------------------------
// Pack an (M, Ktot) fp32 weight tensor (source flat ((m*Ktot+k)*taps + tap))
// into WMMA A-matrix lanes with contiguous k = 16*(lane/16) + j, bf16:
//   dst[(((mt*taps + tap)*kch + kc) * 512) + lane*16 + j]
// ---------------------------------------------------------------------------
__global__ void pack_a_kernel(const float* __restrict__ w,
                              unsigned short* __restrict__ dst,
                              int mtiles, int taps, int kch, int Ktot) {
  int total = mtiles * taps * kch * 512;
  for (int d = blockIdx.x * blockDim.x + threadIdx.x; d < total;
       d += gridDim.x * blockDim.x) {
    int j    = d & 15;
    int lane = (d >> 4) & 31;
    int rest = d >> 9;
    int kc   = rest % kch;  rest /= kch;
    int tap  = rest % taps; rest /= taps;
    int mt   = rest;
    int m  = lane & 15;
    int hi = lane >> 4;
    int k  = j + (hi ? 16 : 0);
    int cin  = kc * 32 + k;
    int cout = mt * 16 + m;
    dst[d] = f2bf(w[((size_t)cout * Ktot + cin) * taps + tap]);
  }
}

// ---------------------------------------------------------------------------
// BN folding: scale = g*rsqrt(v+eps), shift = b - m*scale.
// ---------------------------------------------------------------------------
__global__ void bn_fold_kernel(const float* __restrict__ bn_img,
                               const float* __restrict__ bn_pts,
                               const float* __restrict__ bn_m,
                               float* __restrict__ bnbuf) {
  const float EPS = 1e-3f;
  int i = blockIdx.x * blockDim.x + threadIdx.x;
  if (i < 128) {
    float g = bn_img[i], b = bn_img[128 + i], m = bn_img[256 + i], v = bn_img[384 + i];
    float s = g * rsqrtf(v + EPS);
    bnbuf[i] = s; bnbuf[128 + i] = b - m * s;
  } else if (i < 256) {
    int j = i - 128;
    float g = bn_pts[j], b = bn_pts[128 + j], m = bn_pts[256 + j], v = bn_pts[384 + j];
    float s = g * rsqrtf(v + EPS);
    bnbuf[256 + j] = s; bnbuf[384 + j] = b - m * s;
  } else if (i < 272) {
    int j = i - 256;
    float g = bn_m[j], b = bn_m[16 + j], m = bn_m[32 + j], v = bn_m[48 + j];
    float s = g * rsqrtf(v + EPS);
    bnbuf[512 + j] = s; bnbuf[528 + j] = b - m * s;
  }
}

// ---------------------------------------------------------------------------
// fp32 (C=64, vox) channel-major -> padded bf16 [pvox][64]; halo = 0.
// ---------------------------------------------------------------------------
__global__ void pad_cvt_kernel(const float* __restrict__ src,
                               unsigned short* __restrict__ dst) {
  int idx = blockIdx.x * blockDim.x + threadIdx.x;    // over P1VOX*64
  int c  = idx & (CIN - 1);
  int pv = idx >> 6;
  int pl = pv % P1L; int t = pv / P1L;
  int pw = t % P1W;  int ph = t / P1W;
  int h = ph - 3, w = pw - 3, l = pl - 3;
  unsigned short val = 0;
  if ((unsigned)h < (unsigned)HD && (unsigned)w < (unsigned)WD &&
      (unsigned)l < (unsigned)LD)
    val = f2bf(src[(size_t)c * NVOX + ((h * WD + w) * LD + l)]);
  dst[idx] = val;
}

// ---------------------------------------------------------------------------
// zero-fill (uint4 granularity) for padded feats halo.
// ---------------------------------------------------------------------------
__global__ void zero_kernel(uint4* __restrict__ p, int n4) {
  int i = blockIdx.x * blockDim.x + threadIdx.x;
  if (i < n4) p[i] = make_uint4(0u, 0u, 0u, 0u);
}

// ---------------------------------------------------------------------------
// 7x7x7 conv, pad 3 (pre-padded input), CIN=64 -> COUT=128, BN+ReLU fused.
// One wave: 32 cout (2 A tiles) x 64 vox (4 B sub-tiles) = 8 accumulators;
// K = 343 taps * 2 chunks; per tap 16 WMMAs on 24 b128 loads (1.5 ld/wmma).
// kx fully unrolled so tap deltas fold to load IOFFSETs; branch-free body.
// ---------------------------------------------------------------------------
__global__ void conv1_kernel(const unsigned short* __restrict__ inPad,
                             const unsigned short* __restrict__ wpack,
                             const float* __restrict__ scale,
                             const float* __restrict__ shift,
                             unsigned short* __restrict__ feats,
                             int chanOff) {
  int wave = threadIdx.x >> 5;
  int lane = threadIdx.x & 31;
  int tile = blockIdx.x * (blockDim.x >> 5) + wave;   // 2048 tiles
  int mtp = tile >> 9;           // 0..3 (pair of cout tiles)
  int vt  = tile & 511;          // 64-voxel tile
  int n  = lane & 15;
  int hi = lane >> 4;

  int vh[4], vw[4], vl[4];
  size_t bbase[4];
#pragma unroll
  for (int s = 0; s < 4; ++s) {
    int v = vt * 64 + s * 16 + n;
    int h = v >> 9, w = (v >> 3) & 63, l = v & 7;
    vh[s] = h; vw[s] = w; vl[s] = l;
    bbase[s] = (size_t)((h * P1W + w) * P1L + l) * CIN + hi * 16;
  }

  v8f acc[2][4];
#pragma unroll
  for (int mtl = 0; mtl < 2; ++mtl)
#pragma unroll
    for (int s = 0; s < 4; ++s)
#pragma unroll
      for (int r = 0; r < 8; ++r) acc[mtl][s][r] = 0.f;

  const unsigned short* wbase =
      wpack + ((size_t)(mtp * 2 * 343) << 10) + (lane << 4);  // tap stride 1024

  for (int kz = 0; kz < 7; ++kz) {
    for (int ky = 0; ky < 7; ++ky) {
      int tap0  = (kz * 7 + ky) * 7;
      int toff0 = (kz * P1W + ky) * P1L;
      const unsigned short* ap = wbase + ((size_t)tap0 << 10);
      const unsigned short* bp0 = inPad + (size_t)toff0 * CIN;
      __builtin_prefetch(bp0 + bbase[0] + (size_t)P1L * CIN, 0, 1);
#pragma unroll
      for (int kx = 0; kx < 7; ++kx) {
        v16bf a00 = *(const v16bf*)(ap + kx * 1024);
        v16bf a01 = *(const v16bf*)(ap + kx * 1024 + 512);
        v16bf a10 = *(const v16bf*)(ap + kx * 1024 + MT1OFF);
        v16bf a11 = *(const v16bf*)(ap + kx * 1024 + MT1OFF + 512);
#pragma unroll
        for (int s = 0; s < 4; ++s) {
          const unsigned short* bp = bp0 + bbase[s] + (size_t)kx * CIN;
          v16bf b0 = *(const v16bf*)(bp);
          v16bf b1 = *(const v16bf*)(bp + 32);
          acc[0][s] = wmma_bf16(a00, b0, acc[0][s]);
          acc[0][s] = wmma_bf16(a01, b1, acc[0][s]);
          acc[1][s] = wmma_bf16(a10, b0, acc[1][s]);
          acc[1][s] = wmma_bf16(a11, b1, acc[1][s]);
        }
      }
    }
  }

  // epilogue: BN + ReLU -> padded feats bf16 [pvox][256]
#pragma unroll
  for (int s = 0; s < 4; ++s) {
    size_t pf = (size_t)(((vh[s] + 1) * P2W + (vw[s] + 1)) * P2L + (vl[s] + 1)) * 256;
#pragma unroll
    for (int mtl = 0; mtl < 2; ++mtl)
#pragma unroll
      for (int r = 0; r < 8; ++r) {
        int ch = (mtp * 2 + mtl) * 16 + r + hi * 8;
        float y = acc[mtl][s][r] * scale[ch] + shift[ch];
        y = y > 0.f ? y : 0.f;
        feats[pf + chanOff + ch] = f2bf(y);
      }
  }
}

// ---------------------------------------------------------------------------
// multi_enc: 3x3x3 conv 256->16 (pre-padded feats) + BN + ReLU, then 1x1x1
// conv 16->1 + sigmoid. One wave per 16-voxel tile; halves combined via
// shfl_xor(16).
// ---------------------------------------------------------------------------
__global__ void conv2_vis_kernel(const unsigned short* __restrict__ feats,
                                 const unsigned short* __restrict__ wpack,
                                 const float* __restrict__ scale,
                                 const float* __restrict__ shift,
                                 const float* __restrict__ w2,  // (16,)
                                 float* __restrict__ vis) {
  int wave = threadIdx.x >> 5;
  int lane = threadIdx.x & 31;
  int vt = blockIdx.x * (blockDim.x >> 5) + wave;     // 2048 tiles
  int n  = lane & 15;
  int hi = lane >> 4;
  int v  = vt * 16 + n;
  int h = v >> 9, w = (v >> 3) & 63, l = v & 7;
  size_t bbase = (size_t)((h * P2W + w) * P2L + l) * 256 + hi * 16;

  v8f acc;
#pragma unroll
  for (int r = 0; r < 8; ++r) acc[r] = 0.f;

  for (int kz = 0; kz < 3; ++kz) {
    for (int ky = 0; ky < 3; ++ky) {
      int tap0  = (kz * 3 + ky) * 3;
      int toff0 = (kz * P2W + ky) * P2L;
      const unsigned short* ap =
          wpack + ((size_t)(tap0 * 8) << 9) + (lane << 4);
      const unsigned short* bp0 = feats + bbase + (size_t)toff0 * 256;
#pragma unroll
      for (int kx = 0; kx < 3; ++kx) {
#pragma unroll
        for (int kc = 0; kc < 8; ++kc) {
          v16bf b = *(const v16bf*)(bp0 + (size_t)kx * 256 + kc * 32);
          v16bf a = *(const v16bf*)(ap + (kx * 8 + kc) * 512);
          acc = wmma_bf16(a, b, acc);
        }
      }
    }
  }
  float partial = 0.f;
#pragma unroll
  for (int r = 0; r < 8; ++r) {
    int ch = r + hi * 8;
    float g = acc[r] * scale[ch] + shift[ch];
    g = g > 0.f ? g : 0.f;
    partial += g * w2[ch];
  }
  float tot = partial + __shfl_xor(partial, 16, 32);
  if (hi == 0) vis[v] = 1.f / (1.f + __expf(-tot));
}

// ---------------------------------------------------------------------------
// Fuse img/pts by vis weight, emit xf[vox][128] bf16, compute top-1 gate.
// ---------------------------------------------------------------------------
__global__ void fuse_gate_kernel(const unsigned short* __restrict__ feats,
                                 const float* __restrict__ vis,
                                 const float* __restrict__ w_gate, // (128,3)
                                 unsigned short* __restrict__ xf,
                                 int* __restrict__ gate) {
  int v = blockIdx.x * blockDim.x + threadIdx.x;
  if (v >= NVOX) return;
  int h = v >> 9, w = (v >> 3) & 63, l = v & 7;
  size_t pf = (size_t)((h * P2W + w) * P2L + l + P2CENTER) * 256;
  float vw = vis[v];
  float lg0 = 0.f, lg1 = 0.f, lg2 = 0.f;
  const unsigned short* row = feats + pf;
  for (int c = 0; c < 128; ++c) {
    float fi = bf2f(row[c]);
    float fp = bf2f(row[128 + c]);
    float f = vw * fi + (1.f - vw) * fp;
    xf[(size_t)v * 128 + c] = f2bf(f);
    lg0 += f * w_gate[c * 3 + 0];
    lg1 += f * w_gate[c * 3 + 1];
    lg2 += f * w_gate[c * 3 + 2];
  }
  int best = 0;
  float bl = lg0;
  if (lg1 > bl) { bl = lg1; best = 1; }
  if (lg2 > bl) { bl = lg2; best = 2; }
  gate[v] = best;
}

// ---------------------------------------------------------------------------
// MoE: dense over 3 experts, WMMA fc1 -> ReLU -> LDS relayout -> WMMA fc2,
// one-hot select by gate, write fp32 channel-major output.
// ---------------------------------------------------------------------------
__global__ void moe_kernel(const unsigned short* __restrict__ xf,
                           const unsigned short* __restrict__ fc1p,
                           const unsigned short* __restrict__ fc2p,
                           const float* __restrict__ fc1b,  // (NE,HID)
                           const float* __restrict__ fc2b,  // (NE,COUT)
                           const int* __restrict__ gate,
                           float* __restrict__ out) {
  __shared__ unsigned short hbuf[8][16 * 128];
  int wave = threadIdx.x >> 5;
  int lane = threadIdx.x & 31;
  int vt = blockIdx.x * 8 + wave;        // 2048 tiles, 8 waves/block
  int n  = lane & 15;
  int hi = lane >> 4;
  int v  = vt * 16 + n;
  int gidx = gate[v];

  v16bf Bx[4];
#pragma unroll
  for (int kc = 0; kc < 4; ++kc)
    Bx[kc] = *(const v16bf*)(xf + (size_t)v * 128 + kc * 32 + hi * 16);

  float y[8][8] = {};

  for (int e = 0; e < NE; ++e) {
    __syncthreads();
    const unsigned short* f1 = fc1p + (size_t)e * (8 * 4 * 512);
#pragma unroll
    for (int mt = 0; mt < 8; ++mt) {
      v8f acc;
#pragma unroll
      for (int r = 0; r < 8; ++r)
        acc[r] = fc1b[e * HID + mt * 16 + r + hi * 8];
#pragma unroll
      for (int kc = 0; kc < 4; ++kc) {
        v16bf a = *(const v16bf*)(f1 + (((size_t)(mt * 4 + kc)) << 9) + (lane << 4));
        acc = wmma_bf16(a, Bx[kc], acc);
      }
#pragma unroll
      for (int r = 0; r < 8; ++r) {
        float hv = acc[r] > 0.f ? acc[r] : 0.f;
        hbuf[wave][n * 128 + mt * 16 + r + hi * 8] = f2bf(hv);
      }
    }
    __syncthreads();
    const unsigned short* f2 = fc2p + (size_t)e * (8 * 4 * 512);
#pragma unroll
    for (int mt = 0; mt < 8; ++mt) {
      v8f acc;
#pragma unroll
      for (int r = 0; r < 8; ++r)
        acc[r] = fc2b[e * COUT + mt * 16 + r + hi * 8];
#pragma unroll
      for (int kc = 0; kc < 4; ++kc) {
        v16bf a = *(const v16bf*)(f2 + (((size_t)(mt * 4 + kc)) << 9) + (lane << 4));
        v16bf bh = *(const v16bf*)(&hbuf[wave][n * 128 + kc * 32 + hi * 16]);
        acc = wmma_bf16(a, bh, acc);
      }
      if (gidx == e)
#pragma unroll
        for (int r = 0; r < 8; ++r) y[mt][r] += acc[r];
    }
  }
#pragma unroll
  for (int mt = 0; mt < 8; ++mt)
#pragma unroll
    for (int r = 0; r < 8; ++r)
      out[(size_t)(mt * 16 + r + hi * 8) * NVOX + v] = y[mt][r];
}

// ---------------------------------------------------------------------------
// Host-side launch
// ---------------------------------------------------------------------------
extern "C" void kernel_launch(void* const* d_in, const int* in_sizes, int n_in,
                              void* d_out, int out_size, void* d_ws, size_t ws_size,
                              hipStream_t stream) {
  const float* img    = (const float*)d_in[0];
  const float* pts    = (const float*)d_in[1];
  const float* w_img  = (const float*)d_in[2];
  const float* w_pts  = (const float*)d_in[3];
  const float* bn_img = (const float*)d_in[4];
  const float* bn_pts = (const float*)d_in[5];
  const float* w_m1   = (const float*)d_in[6];
  const float* bn_m   = (const float*)d_in[7];
  const float* w_m2   = (const float*)d_in[8];
  const float* w_gate = (const float*)d_in[9];
  const float* fc1_w  = (const float*)d_in[10];
  const float* fc1_b  = (const float*)d_in[11];
  const float* fc2_w  = (const float*)d_in[12];
  const float* fc2_b  = (const float*)d_in[13];
  float* out = (float*)d_out;

  // carve workspace
  char* base = (char*)d_ws;
  size_t off = 0;
  auto carve = [&](size_t bytes) -> char* {
    off = (off + 255) & ~(size_t)255;
    char* p = base + off;
    off += bytes;
    return p;
  };
  unsigned short* wpi  = (unsigned short*)carve((size_t)8 * 343 * 2 * 512 * 2);
  unsigned short* wpp  = (unsigned short*)carve((size_t)8 * 343 * 2 * 512 * 2);
  unsigned short* wm1  = (unsigned short*)carve((size_t)1 * 27 * 8 * 512 * 2);
  unsigned short* wf1  = (unsigned short*)carve((size_t)NE * 8 * 4 * 512 * 2);
  unsigned short* wf2  = (unsigned short*)carve((size_t)NE * 8 * 4 * 512 * 2);
  float*          bnb  = (float*)carve(544 * 4);
  unsigned short* inPi = (unsigned short*)carve((size_t)P1VOX * CIN * 2);
  unsigned short* inPp = (unsigned short*)carve((size_t)P1VOX * CIN * 2);
  unsigned short* fts  = (unsigned short*)carve((size_t)P2VOX * 256 * 2);
  float*          vis  = (float*)carve((size_t)NVOX * 4);
  unsigned short* xf   = (unsigned short*)carve((size_t)NVOX * 128 * 2);
  int*            gate = (int*)carve((size_t)NVOX * 4);
  (void)ws_size; (void)in_sizes; (void)n_in; (void)out_size;

  // stage 0: weight packing + BN folding + padded input convert + feats zero
  pack_a_kernel<<<1024, 256, 0, stream>>>(w_img, wpi, 8, 343, 2, CIN);
  pack_a_kernel<<<1024, 256, 0, stream>>>(w_pts, wpp, 8, 343, 2, CIN);
  pack_a_kernel<<<256, 256, 0, stream>>>(w_m1, wm1, 1, 27, 8, 256);
  for (int e = 0; e < NE; ++e) {
    pack_a_kernel<<<64, 256, 0, stream>>>(fc1_w + (size_t)e * HID * COUT,
                                          wf1 + (size_t)e * 8 * 4 * 512, 8, 1, 4, COUT);
    pack_a_kernel<<<64, 256, 0, stream>>>(fc2_w + (size_t)e * COUT * HID,
                                          wf2 + (size_t)e * 8 * 4 * 512, 8, 1, 4, HID);
  }
  bn_fold_kernel<<<2, 256, 0, stream>>>(bn_img, bn_pts, bn_m, bnb);
  {
    int total = P1VOX * CIN;                       // 4,390,400
    pad_cvt_kernel<<<(total + 255) / 256, 256, 0, stream>>>(img, inPi);
    pad_cvt_kernel<<<(total + 255) / 256, 256, 0, stream>>>(pts, inPp);
    int n4 = (P2VOX * 256 * 2) / 16;               // 1,393,920 uint4
    zero_kernel<<<(n4 + 255) / 256, 256, 0, stream>>>((uint4*)fts, n4);
  }

  // stage 1: the two 7x7x7 convs (2048 wave-tiles each, 8 waves/block)
  conv1_kernel<<<256, 256, 0, stream>>>(inPi, wpi, bnb + 0, bnb + 128, fts, 0);
  conv1_kernel<<<256, 256, 0, stream>>>(inPp, wpp, bnb + 256, bnb + 384, fts, 128);

  // stage 2: multi_enc conv + 1x1x1 + sigmoid -> vis
  conv2_vis_kernel<<<256, 256, 0, stream>>>(fts, wm1, bnb + 512, bnb + 528, w_m2, vis);

  // stage 3: fuse + gate
  fuse_gate_kernel<<<NVOX / 256, 256, 0, stream>>>(fts, vis, w_gate, xf, gate);

  // stage 4: MoE
  moe_kernel<<<256, 256, 0, stream>>>(xf, wf1, wf2, fc1_b, fc2_b, gate, out);
}
